// MonarchAttentionPerformer_27522150433136
// MI455X (gfx1250) — matrix operand order repaired and checked by reading, not compiled, not run on hardware
//
// MonarchAttentionPerformer for MI455X (gfx1250, wave32, WMMA).
// GEMM core: 256 thr = 8 waves, block tile 128x64, wave tile 32x32
// (2x2 register blocking -> 4 independent v_wmma per K-step), K step 32,
// double-buffered LDS with ASYNC global->LDS staging
// (global_load_async_to_lds_b128 + s_wait_asynccnt, one s_barrier per step),
// global_prefetch_b8 two slabs ahead.
// Pipeline: cvt fp32->f16; monarch kqv (fused permute+bias); row norms;
// prm_exp GEMMs (fused exp(acc-xd)/sqrt(M)); kp colsums; D = qp . sum kp;
// kptv = V^T KP; y = QP kptv^T / (D+eps) (fused divide); monarch proj -> f32.
// ~310 MB f16 intermediates in d_ws; pipeline is HBM-bandwidth-bound at
// 23.3 TB/s (~135 GFLOP total), so 16-bit traffic + fused epilogues dominate.

#include <hip/hip_runtime.h>
#include <hip/hip_bf16.h>
#include <math.h>

typedef _Float16 f16;
typedef _Float16 v16h __attribute__((ext_vector_type(16)));
typedef _Float16 v8h  __attribute__((ext_vector_type(8)));
typedef float    v8f  __attribute__((ext_vector_type(8)));

#define EP_F16     0
#define EP_BIASF16 1
#define EP_BIASF32 2
#define EP_EXPF16  3
#define EP_DIVF16  4

#define LDS_STRIDE 48   // halves; rows stay 16B-aligned, staggers banks

// ---- CDNA5 async global->LDS copy (ASYNCcnt-tracked DMA, bypasses VGPRs) ----
// LDS byte address = low 32 bits of the generic shared pointer (flat aperture
// rule); global address in a 64-bit VGPR pair.
__device__ __forceinline__ void async_cp_b128(const f16* g, f16* l) {
  asm volatile("global_load_async_to_lds_b128 %0, %1, off"
               :: "v"((unsigned)(size_t)l), "v"((unsigned long long)(size_t)g)
               : "memory");
}
__device__ __forceinline__ void async_fence() {
  asm volatile("s_wait_asynccnt 0x0" ::: "memory");
}

// Generic tiled WMMA GEMM: C(row,col) = sum_k A(row,k)*B(k,col), epilogue EP.
// TA: A(m,k) = A[k*lda+m] if 1 else A[m*lda+k]
// TB: B(k,n) = B[n*ldb+k] if 1 else B[k*ldb+n]
// Store index: z*sCz + row*ldc + col*colStride + z*colOffZ  (monarch permute)
template<int EP, int TA, int TB>
__global__ __launch_bounds__(256)
void gemm_wmma(const f16* __restrict__ A, int lda, long long sAz,
               const f16* __restrict__ B, int ldb, long long sBz,
               void* __restrict__ Cv, int ldc, long long sCz,
               int colStride, int colOffZ,
               const float* __restrict__ bias,
               const float* __restrict__ rowvec, float scale, int rowZ,
               int Kdim)
{
  __shared__ f16 As[2][128 * LDS_STRIDE];  // 128 rows x 32 K  (12 KB each)
  __shared__ f16 Bs[2][64 * LDS_STRIDE];   // [n][k] transposed (6 KB each)

  const int z = blockIdx.z;
  A += (long long)z * sAz;
  B += (long long)z * sBz;

  const int tid   = threadIdx.x;
  const int lane  = tid & 31;
  const int wave  = tid >> 5;          // 8 waves
  const int waveM = wave & 3;          // 4 row-waves  (32 rows each)
  const int waveN = wave >> 2;         // 2 col-waves  (32 cols each)
  const int l16   = lane & 15;
  const int lHi   = lane >> 4;
  const int tileM = blockIdx.x * 128;
  const int tileN = blockIdx.y * 64;

  // ---- cooperative global -> LDS stage of one K-slab (k0..k0+31) ----
  auto stage = [&](int k0, f16* Asb, f16* Bsb) {
    if (TA == 0) {
      const int row = tid >> 1, ch = (tid & 1) * 16;     // 16 halves/thread
      const f16* src = A + (size_t)(tileM + row) * lda + k0 + ch;
      f16* dst = &Asb[row * LDS_STRIDE + ch];
      async_cp_b128(src,     dst);                        // ASYNCcnt DMA
      async_cp_b128(src + 8, dst + 8);
    } else {
      const int kk = tid >> 3, cn = (tid & 7) * 16;
      const f16* src = A + (size_t)(k0 + kk) * lda + tileM + cn;
      v8h v0 = *(const v8h*)(src);
      v8h v1 = *(const v8h*)(src + 8);
      #pragma unroll
      for (int j = 0; j < 8; ++j) {
        Asb[(cn + j) * LDS_STRIDE + kk]     = v0[j];
        Asb[(cn + 8 + j) * LDS_STRIDE + kk] = v1[j];
      }
    }
    if (TB == 1) {
      const int n = tid >> 2, ck = (tid & 3) * 8;        // 8 halves/thread
      async_cp_b128(B + (size_t)(tileN + n) * ldb + k0 + ck,
                    &Bsb[n * LDS_STRIDE + ck]);
    } else {
      const int kk = tid >> 3, cn = (tid & 7) * 8;
      v8h v0 = *(const v8h*)(B + (size_t)(k0 + kk) * ldb + tileN + cn);
      #pragma unroll
      for (int j = 0; j < 8; ++j) Bsb[(cn + j) * LDS_STRIDE + kk] = v0[j];
    }
  };

  // ---- fragment fetch per ISA 7.12.2 (16-bit A 16x32 / B 32x16) ----
  auto fragA = [&](const f16* Asb, int mi) {
    const f16* arow = &Asb[(waveM * 32 + mi * 16 + l16) * LDS_STRIDE];
    v16h a;
    ((v8h*)&a)[0] = *(const v8h*)(arow + lHi * 8);       // K = lHi*8 + 0..7
    ((v8h*)&a)[1] = *(const v8h*)(arow + 16 + lHi * 8);  // K = 16+lHi*8+0..7
    return a;
  };
  auto fragB = [&](const f16* Bsb, int ni) {
    const f16* bcol =
        &Bsb[(waveN * 32 + ni * 16 + l16) * LDS_STRIDE + lHi * 8];
    v16h b;
    ((v8h*)&b)[0] = *(const v8h*)(bcol);                 // K = lHi*8 + 0..7
    ((v8h*)&b)[1] = *(const v8h*)(bcol + 16);            // K = 16+lHi*8+0..7
    return b;
  };

  v8f acc[2][2] = {};

  stage(0, As[0], Bs[0]);
  async_fence();
  __syncthreads();

  const int nk = Kdim >> 5;
  for (int it = 0; it < nk; ++it) {
    const int cur = it & 1;
    if (it + 1 < nk) {
      stage((it + 1) * 32, As[cur ^ 1], Bs[cur ^ 1]);    // overlaps WMMAs below
      if (it + 2 < nk) {   // lowers to global_prefetch_b8
        const int kpf = (it + 2) * 32;
        const f16* pa = TA ? A + (size_t)(kpf + (tid >> 3)) * lda + tileM
                           : A + (size_t)(tileM + (tid >> 1)) * lda + kpf;
        const f16* pb = TB ? B + (size_t)(tileN + (tid >> 2)) * ldb + kpf
                           : B + (size_t)(kpf + (tid >> 3)) * ldb + tileN;
        __builtin_prefetch(pa, 0, 1);
        __builtin_prefetch(pb, 0, 1);
      }
    }
    v16h a0 = fragA(As[cur], 0);
    v16h a1 = fragA(As[cur], 1);
    v16h b0 = fragB(Bs[cur], 0);
    v16h b1 = fragB(Bs[cur], 1);
    acc[0][0] = __builtin_amdgcn_wmma_f32_16x16x32_f16(false, a0, false, b0,
                                                       (short)0, acc[0][0], false, false);
    acc[0][1] = __builtin_amdgcn_wmma_f32_16x16x32_f16(false, a0, false, b1,
                                                       (short)0, acc[0][1], false, false);
    acc[1][0] = __builtin_amdgcn_wmma_f32_16x16x32_f16(false, a1, false, b0,
                                                       (short)0, acc[1][0], false, false);
    acc[1][1] = __builtin_amdgcn_wmma_f32_16x16x32_f16(false, a1, false, b1,
                                                       (short)0, acc[1][1], false, false);
    async_fence();          // next slab fully resident in LDS
    __syncthreads();        // publish to all 8 waves
  }

  // ---- epilogue: C/D layout VGPR i -> row = i + 8*laneHi ----
  #pragma unroll
  for (int mi = 0; mi < 2; ++mi) {
    #pragma unroll
    for (int ni = 0; ni < 2; ++ni) {
      const int rowB = tileM + waveM * 32 + mi * 16 + lHi * 8;
      const int col  = tileN + waveN * 32 + ni * 16 + l16;
      const int colIdx = col * colStride + z * colOffZ;
      #pragma unroll
      for (int i = 0; i < 8; ++i) {
        const int row = rowB + i;
        float v = acc[mi][ni][i];
        if (EP == EP_BIASF16 || EP == EP_BIASF32) v += bias[colIdx];
        if (EP == EP_EXPF16) v = __expf(v - rowvec[(size_t)z * rowZ + row]) * scale;
        if (EP == EP_DIVF16) v = v / (rowvec[(size_t)z * rowZ + row] + 1e-8f);
        if (EP == EP_BIASF32)
          ((float*)Cv)[(size_t)z * sCz + (size_t)row * ldc + colIdx] = v;
        else
          ((f16*)Cv)[(size_t)z * sCz + (size_t)row * ldc + colIdx] = (f16)v;
      }
    }
  }
}

__global__ void cvt_f32_f16(const float* __restrict__ src, f16* __restrict__ dst,
                            long long n) {
  long long i = (long long)blockIdx.x * 256 + threadIdx.x;
  if (i < n) dst[i] = (f16)src[i];
}

// out[row] = 0.5 * sum_d X[row*ld + d]^2   (one wave32 per row)
__global__ void rownorm_half(const f16* __restrict__ X, int ld, int dim,
                             float* __restrict__ out) {
  const int row  = blockIdx.x * 4 + (threadIdx.x >> 5);
  const int lane = threadIdx.x & 31;
  const f16* xr = X + (size_t)row * ld;
  float s = 0.f;
  for (int j = lane; j < dim; j += 32) { float v = (float)xr[j]; s += v * v; }
  for (int off = 16; off; off >>= 1) s += __shfl_down(s, off, 32);
  if (lane == 0) out[row] = 0.5f * s;
}

// ksum[b, m] = sum_t kp[b, t, m]
__global__ void colsum_half(const f16* __restrict__ kp, float* __restrict__ ksum) {
  const int b = blockIdx.y;
  const int m = blockIdx.x * 128 + threadIdx.x;           // grid.x = 3 -> 384
  const f16* p = kp + (size_t)b * 4096 * 384 + m;
  float s = 0.f;
  for (int t = 0; t < 4096; ++t) s += (float)p[(size_t)t * 384];
  ksum[b * 384 + m] = s;
}

// D[row] = sum_m qp[row,m] * ksum[b,m]   (one wave32 per row)
__global__ void dvec_half(const f16* __restrict__ qp, const float* __restrict__ ksum,
                          float* __restrict__ Dv) {
  const int row  = blockIdx.x * 4 + (threadIdx.x >> 5);
  const int lane = threadIdx.x & 31;
  const int b = row >> 12;                                 // /4096
  const f16* q = qp + (size_t)row * 384;
  const float* ks = ksum + b * 384;
  float s = 0.f;
  for (int m = lane; m < 384; m += 32) s += (float)q[m] * ks[m];
  for (int off = 16; off; off >>= 1) s += __shfl_down(s, off, 32);
  if (lane == 0) Dv[row] = s;
}

extern "C" void kernel_launch(void* const* d_in, const int* in_sizes, int n_in,
                              void* d_out, int out_size, void* d_ws, size_t ws_size,
                              hipStream_t stream)
{
  (void)in_sizes; (void)n_in; (void)out_size; (void)ws_size;
  const float* x      = (const float*)d_in[0];
  const float* kqv_w1 = (const float*)d_in[1];   // (4,192,192)
  const float* kqv_w2 = (const float*)d_in[2];   // (4,576,192)
  const float* kqv_b  = (const float*)d_in[3];   // (2304,)
  const float* p_w1   = (const float*)d_in[4];   // (4,192,192)
  const float* p_w2   = (const float*)d_in[5];   // (4,192,192)
  const float* p_b    = (const float*)d_in[6];   // (768,)
  const float* w      = (const float*)d_in[7];   // (384,768)

  const int Bb = 8, T = 4096, N = Bb * T, D = 768, M3 = 2304, MF = 384;

  char* base = (char*)d_ws;
  size_t off = 0;
  auto alloc = [&](size_t bytes) -> char* {
    char* p = base + off;
    off = (off + bytes + 255) & ~(size_t)255;
    return p;
  };
  f16* xh    = (f16*)alloc((size_t)N * D * 2);        // also reused as y (f16)
  f16* o1h   = (f16*)alloc((size_t)N * D * 2);        // monarch intermediate
  f16* kqvh  = (f16*)alloc((size_t)N * M3 * 2);       // [k|q|v] f16
  f16* kph   = (f16*)alloc((size_t)N * MF * 2);
  f16* qph   = (f16*)alloc((size_t)N * MF * 2);
  f16* kptvh = (f16*)alloc((size_t)Bb * D * MF * 2);
  f16* w1h   = (f16*)alloc((size_t)147456 * 2);
  f16* w2h   = (f16*)alloc((size_t)442368 * 2);
  f16* wh    = (f16*)alloc((size_t)294912 * 2);
  f16* pw1h  = (f16*)alloc((size_t)147456 * 2);
  f16* pw2h  = (f16*)alloc((size_t)147456 * 2);
  float* xdk  = (float*)alloc((size_t)N * 4);
  float* xdq  = (float*)alloc((size_t)N * 4);
  float* Dv   = (float*)alloc((size_t)N * 4);
  float* ksum = (float*)alloc((size_t)Bb * MF * 4);
  f16* yh = xh;

  const float scaleM = 1.0f / sqrtf((float)MF);

  // ---- fp32 -> f16 conversions ----
  auto cvt = [&](const float* s, f16* d, long long n) {
    cvt_f32_f16<<<(unsigned)((n + 255) / 256), 256, 0, stream>>>(s, d, n);
  };
  cvt(x, xh, (long long)N * D);
  cvt(kqv_w1, w1h, 147456);
  cvt(kqv_w2, w2h, 442368);
  cvt(w, wh, 294912);
  cvt(p_w1, pw1h, 147456);
  cvt(p_w2, pw2h, 147456);

  // ---- monarch kqv: stage1  o1[row, q*4+k] = x_blk . w1^T ----
  gemm_wmma<EP_F16, 0, 1><<<dim3(256, 3, 4), 256, 0, stream>>>(
      xh, D, 192LL, w1h, 192, 192LL * 192,
      o1h, D, 0LL, 4, 1, nullptr, nullptr, 0.f, 0, 192);
  // ---- stage2  kqv[row, s*4+l] = o1_blk . w2^T + b ----
  gemm_wmma<EP_BIASF16, 0, 1><<<dim3(256, 9, 4), 256, 0, stream>>>(
      o1h, D, 192LL, w2h, 192, 576LL * 192,
      kqvh, M3, 0LL, 4, 1, kqv_b, nullptr, 0.f, 0, 192);

  // ---- xd = 0.5||k||^2 , 0.5||q||^2 ----
  rownorm_half<<<N / 4, 128, 0, stream>>>(kqvh,     M3, D, xdk);
  rownorm_half<<<N / 4, 128, 0, stream>>>(kqvh + D, M3, D, xdq);

  // ---- prm_exp: kp/qp = exp(k.w^T - xd)/sqrt(M) ----
  gemm_wmma<EP_EXPF16, 0, 1><<<dim3(256, 6, 1), 256, 0, stream>>>(
      kqvh, M3, 0LL, wh, D, 0LL,
      kph, MF, 0LL, 1, 0, nullptr, xdk, scaleM, 0, D);
  gemm_wmma<EP_EXPF16, 0, 1><<<dim3(256, 6, 1), 256, 0, stream>>>(
      kqvh + D, M3, 0LL, wh, D, 0LL,
      qph, MF, 0LL, 1, 0, nullptr, xdq, scaleM, 0, D);

  // ---- ksum[b,m] , D[row] ----
  colsum_half<<<dim3(3, Bb), 128, 0, stream>>>(kph, ksum);
  dvec_half<<<N / 4, 128, 0, stream>>>(qph, ksum, Dv);

  // ---- kptv[b,n,m] = sum_t v[b,t,n] * kp[b,t,m]  (A transposed, K=4096) ----
  gemm_wmma<EP_F16, 1, 0><<<dim3(6, 6, 8), 256, 0, stream>>>(
      kqvh + 2 * D, M3, (long long)T * M3, kph, MF, (long long)T * MF,
      kptvh, MF, (long long)D * MF, 1, 0, nullptr, nullptr, 0.f, 0, T);

  // ---- y[b,t,n] = (qp . kptv[b,n,:]) / (D + eps)  (B transposed, K=384) ----
  gemm_wmma<EP_DIVF16, 0, 1><<<dim3(32, 12, 8), 256, 0, stream>>>(
      qph, MF, (long long)T * MF, kptvh, MF, (long long)D * MF,
      yh, D, (long long)T * D, 1, 0, nullptr, Dv, 0.f, T, MF);

  // ---- monarch proj: stage1 + stage2 (bias, fp32 out) ----
  gemm_wmma<EP_F16, 0, 1><<<dim3(256, 3, 4), 256, 0, stream>>>(
      yh, D, 192LL, pw1h, 192, 192LL * 192,
      o1h, D, 0LL, 4, 1, nullptr, nullptr, 0.f, 0, 192);
  gemm_wmma<EP_BIASF32, 0, 1><<<dim3(256, 3, 4), 256, 0, stream>>>(
      o1h, D, 192LL, pw2h, 192, 192LL * 192,
      d_out, D, 0LL, 4, 1, p_b, nullptr, 0.f, 0, 192);
}
